// AttentionBlock_31842887532919
// MI455X (gfx1250) — matrix operand order repaired and checked by reading, not compiled, not run on hardware
//
#include <hip/hip_runtime.h>
#include <hip/hip_bf16.h>

// ---------------------------------------------------------------------------
// AttentionBlock for MI455X (gfx1250): wave32 + v_wmma_f32_16x16x32_f16
// Dims fixed by reference: B=2, QS=KS=2048, E=2048, H=16, D=128
// ---------------------------------------------------------------------------

typedef __attribute__((ext_vector_type(16))) _Float16 v16h;
typedef __attribute__((ext_vector_type(8)))  _Float16 v8h;
typedef __attribute__((ext_vector_type(8)))  float    v8f;
typedef __attribute__((ext_vector_type(4)))  int      v4i;

constexpr int Bn = 2;
constexpr int S  = 2048;   // QS == KS
constexpr int E  = 2048;
constexpr int H  = 16;
constexpr int D  = 128;
constexpr float EPS        = 1e-6f;
constexpr float BIG_NEG    = -3.0e38f;
constexpr float INV_SQRT_D = 0.088388347648318447f; // 1/sqrt(128)

// gfx1250 async global->LDS copy (ASYNCcnt-tracked), guarded so the build can
// never break if this toolchain doesn't expose the builtins.
#if defined(__gfx1250__) && \
    __has_builtin(__builtin_amdgcn_global_load_async_to_lds_b128) && \
    __has_builtin(__builtin_amdgcn_s_wait_asynccnt)
#define ASYNC_LDS 1
#else
#define ASYNC_LDS 0
#endif

#if ASYNC_LDS
typedef __attribute__((address_space(1))) v4i as1_v4i;  // global int4
typedef __attribute__((address_space(3))) v4i as3_v4i;  // LDS int4
#endif

__device__ __forceinline__ void copy16B_to_lds(const _Float16* g, _Float16* l) {
#if ASYNC_LDS
  __builtin_amdgcn_global_load_async_to_lds_b128(
      (as1_v4i*)(void*)g, (as3_v4i*)(void*)l, 0, 0);
#else
  *(v8h*)l = *(const v8h*)g;
#endif
}

__device__ __forceinline__ void async_fence_barrier() {
#if ASYNC_LDS
  __builtin_amdgcn_s_wait_asynccnt(0);   // s_wait_asynccnt 0
#endif
  __syncthreads();                       // waits DScnt, then s_barrier
}

__device__ __forceinline__ v8f wmma_f16(v16h a, v16h b, v8f c) {
  // D = A(16x32) * B(32x16) + C, f32 accumulate
  return __builtin_amdgcn_wmma_f32_16x16x32_f16(
      /*neg_a=*/false, a, /*neg_b=*/false, b,
      /*c_mod=*/(short)0, c, /*reuse_a=*/false, /*reuse_b=*/false);
}

// A-fragment loader: A is row-major M x K, load 16x32 tile at p0 (row stride ld).
// Lane layout: m = lane&15; hi = lane>>4; halves 0..7 -> K = hi*8+j,
// halves 8..15 -> K = 16+hi*8+j  (two aligned 16-byte loads per lane).
__device__ __forceinline__ v16h load_a_frag(const _Float16* __restrict__ p0, int ld) {
  int lane = threadIdx.x & 31;
  int m = lane & 15, hi = lane >> 4;
  const _Float16* p = p0 + (size_t)m * ld + hi * 8;
  v8h x0 = *(const v8h*)p;
  v8h x1 = *(const v8h*)(p + 16);
  v16h r;
#pragma unroll
  for (int i = 0; i < 8; ++i) { r[i] = x0[i]; r[i + 8] = x1[i]; }
  return r;
}

// ---------------------------------------------------------------------------
// Kernel 1: qi = f16(q + q_pos), ki = f16(kv + kv_pos), vi = f16(kv)
// ---------------------------------------------------------------------------
__global__ __launch_bounds__(256) void prep_inputs(
    const float* __restrict__ q, const float* __restrict__ kv,
    const float* __restrict__ qp, const float* __restrict__ kp,
    _Float16* __restrict__ qi, _Float16* __restrict__ ki,
    _Float16* __restrict__ vi) {
  size_t i = (size_t)blockIdx.x * blockDim.x + threadIdx.x;
  qi[i] = (_Float16)(q[i] + qp[i]);
  float k = kv[i];
  ki[i] = (_Float16)(k + kp[i]);
  vi[i] = (_Float16)k;
}

__global__ __launch_bounds__(256) void f32_to_f16(
    const float* __restrict__ in, _Float16* __restrict__ out) {
  size_t i = (size_t)blockIdx.x * blockDim.x + threadIdx.x;
  out[i] = (_Float16)in[i];
}

// ---------------------------------------------------------------------------
// Kernel 2: generic f16 GEMM, C(f32) = A(MxK) * B(KxN), all row-major.
// Block 256 threads = 8 waves; wave tile 32x64, block tile 128x128.
// B k-tile (32x128) double-buffered in LDS via async global->LDS copies.
// ---------------------------------------------------------------------------
constexpr int BROW = 136;  // LDS row stride in halves (272B: 16B-aligned, skews banks)

__global__ __launch_bounds__(256) void gemm_f16f32(
    const _Float16* __restrict__ A, const _Float16* __restrict__ Bm,
    float* __restrict__ C, int M, int N, int K) {
  __shared__ _Float16 Bs[2][32 * BROW];

  int tid  = threadIdx.x;
  int lane = tid & 31;
  int w    = tid >> 5;
  int wm = w & 3, wn = w >> 2;
  int row0 = blockIdx.x * 128 + wm * 32;
  int colg = blockIdx.y * 128;
  int colw = wn * 64;

  // B-tile copy assignment: 256 threads x 2 x 16B covers 32x128 halves
  int crow = tid >> 4;          // 0..15  (and +16)
  int ccol = (tid & 15) * 8;    // 0..120

  // prologue: stage k-tile 0 into buffer 0
  {
    const _Float16* src = Bm + (size_t)crow * N + colg + ccol;
    copy16B_to_lds(src,                  &Bs[0][crow * BROW + ccol]);
    copy16B_to_lds(src + (size_t)16 * N, &Bs[0][(crow + 16) * BROW + ccol]);
  }
  async_fence_barrier();

  v8f acc[2][4] = {};
  const _Float16* a0row = A + (size_t)row0 * K;
  const _Float16* a1row = A + (size_t)(row0 + 16) * K;

  const int nk = K / 32;
  for (int it = 0; it < nk; ++it) {
    int k   = it * 32;
    int cur = it & 1, nxt = cur ^ 1;

    // stage next k-tile while computing on the current one
    if (it + 1 < nk) {
      const _Float16* src = Bm + (size_t)(k + 32 + crow) * N + colg + ccol;
      copy16B_to_lds(src,                  &Bs[nxt][crow * BROW + ccol]);
      copy16B_to_lds(src + (size_t)16 * N, &Bs[nxt][(crow + 16) * BROW + ccol]);
    }

    __builtin_prefetch(a0row + k + 64, 0, 1);   // global_prefetch_b8
    v16h a0 = load_a_frag(a0row + k, K);
    v16h a1 = load_a_frag(a1row + k, K);

    const _Float16* brow = &Bs[cur][(lane >> 4) * 8 * BROW + colw + (lane & 15)];  // lane&15 = n column, elements = k rows
#pragma unroll
    for (int t = 0; t < 4; ++t) {
      v16h b;
#pragma unroll
      for (int i = 0; i < 8; ++i) { b[i] = brow[i * BROW + t * 16]; b[i + 8] = brow[(16 + i) * BROW + t * 16]; }
      acc[0][t] = wmma_f16(a0, b, acc[0][t]);
      acc[1][t] = wmma_f16(a1, b, acc[1][t]);
    }
    async_fence_barrier();
  }

  int n = lane & 15, hi = lane >> 4;
#pragma unroll
  for (int i = 0; i < 2; ++i)
#pragma unroll
    for (int t = 0; t < 4; ++t)
#pragma unroll
      for (int r = 0; r < 8; ++r)
        C[(size_t)(row0 + i * 16 + r + hi * 8) * N + colg + colw + t * 16 + n]
            = acc[i][t][r];
}

// ---------------------------------------------------------------------------
// Kernel 3: per-head LayerNorm over D=128 (no bias). One wave per (b,s,h) row.
// transpose==0: out[b,h,s,d]   (for Q; extra = 1/sqrt(D) folded in)
// transpose==1: out[b,h,d,s]   (pre-transposed K for the QK^T B-fragment)
// ---------------------------------------------------------------------------
__global__ __launch_bounds__(128) void ln_head(
    const float* __restrict__ X,      // [B*S*H, D] == [b,s,h,d]
    const float* __restrict__ scale,  // [D]
    _Float16* __restrict__ out, float extra, int transpose) {
  int lane = threadIdx.x & 31;
  int w    = threadIdx.x >> 5;
  int row  = blockIdx.x * 4 + w;      // (b*S + s)*H + h
  const float* x = X + (size_t)row * D;

  float v[4];
#pragma unroll
  for (int j = 0; j < 4; ++j) v[j] = x[lane + j * 32];

  float s = v[0] + v[1] + v[2] + v[3];
#pragma unroll
  for (int d = 1; d < 32; d <<= 1) s += __shfl_xor(s, d, 32);
  float mu = s * (1.0f / D);

  float sv = 0.f;
#pragma unroll
  for (int j = 0; j < 4; ++j) { float t = v[j] - mu; sv += t * t; }
#pragma unroll
  for (int d = 1; d < 32; d <<= 1) sv += __shfl_xor(sv, d, 32);
  float rstd = rsqrtf(sv * (1.0f / D) + EPS);

  int b = row / (S * H);
  int rem = row % (S * H);
  int sidx = rem / H;
  int h = rem % H;
#pragma unroll
  for (int j = 0; j < 4; ++j) {
    int d = lane + j * 32;
    float y = (v[j] - mu) * rstd * scale[d] * extra;
    size_t off = transpose
        ? (((size_t)(b * H + h) * D + d) * S + sidx)
        : (((size_t)(b * H + h) * S + sidx) * D + d);
    out[off] = (_Float16)y;
  }
}

// ---------------------------------------------------------------------------
// Kernel 4: permute V f32 [b,s,h,d] -> f16 [b,h,s,d]
// ---------------------------------------------------------------------------
__global__ __launch_bounds__(256) void permute_v(
    const float* __restrict__ X, _Float16* __restrict__ out) {
  size_t i = (size_t)blockIdx.x * blockDim.x + threadIdx.x;
  int d = (int)(i % D); size_t t = i / D;
  int h = (int)(t % H); t /= H;
  int s = (int)(t % S); int b = (int)(t / S);
  out[(((size_t)b * H + h) * S + s) * D + d] = (_Float16)X[i];
}

// ---------------------------------------------------------------------------
// Kernel 5: flash attention. Block = 128 threads = 4 waves; each wave owns a
// 16-query strip of one (b,h) and streams keys 32 at a time.
//   qn : [B,H,S,D] f16 (already scaled by 1/sqrt(D))
//   ktn: [B,H,D,S] f16 (K transposed -> contiguous B-fragments)
//   vn : [B,H,S,D] f16
//   mask: [B,S,S] bool (1 byte)
//   out: [B,S,H*D] f16 (attention output, feeds final projection GEMM)
// ---------------------------------------------------------------------------
__global__ __launch_bounds__(128) void flash_attn(
    const _Float16* __restrict__ qn, const _Float16* __restrict__ ktn,
    const _Float16* __restrict__ vn, const unsigned char* __restrict__ mask,
    _Float16* __restrict__ out) {
  __shared__ _Float16 plds[4][16 * 40];   // per-wave 16x32 P staging, 16B-aligned rows

  int lane = threadIdx.x & 31;
  int w    = threadIdx.x >> 5;
  int bh   = blockIdx.x >> 5;             // 32 q-blocks of 64 rows each
  int qblk = blockIdx.x & 31;
  int b = bh >> 4, h = bh & 15;
  int q0 = qblk * 64 + w * 16;
  int n = lane & 15, hi = lane >> 4;

  const _Float16* qbase  = qn  + ((size_t)bh * S + q0) * D;
  const _Float16* ktbase = ktn + (size_t)bh * D * S;
  const _Float16* vbase  = vn  + (size_t)bh * S * D;
  const unsigned char* mbase = mask + (size_t)b * S * S;
  _Float16* pl = plds[w];

  // Q fragments are loop-invariant: preload all 4 d-chunks (16x32 each)
  v16h qf[4];
#pragma unroll
  for (int c = 0; c < 4; ++c) qf[c] = load_a_frag(qbase + c * 32, D);

  v8f o[8] = {};
  float mi[8], li[8];
#pragma unroll
  for (int r = 0; r < 8; ++r) { mi[r] = BIG_NEG; li[r] = 0.f; }

  for (int k0 = 0; k0 < S; k0 += 32) {
    // ---- S = Q * K^T  (two 16x16 tiles: keys [k0,k0+16) and [k0+16,k0+32))
    v8f s0 = {}, s1 = {};
#pragma unroll
    for (int c = 0; c < 4; ++c) {
      const _Float16* kp = ktbase + (size_t)(c * 32 + hi * 8) * S + k0 + n;
      v16h b0, b1;
#pragma unroll
      for (int i = 0; i < 8; ++i) {
        b0[i] = kp[(size_t)i * S];      b0[i + 8] = kp[(size_t)(16 + i) * S];
        b1[i] = kp[(size_t)i * S + 16]; b1[i + 8] = kp[(size_t)(16 + i) * S + 16];
      }
      s0 = wmma_f16(qf[c], b0, s0);
      s1 = wmma_f16(qf[c], b1, s1);
    }

    // ---- mask + online softmax (row m = r + hi*8 lives in 16-lane half)
    float p0[8], p1[8], alpha[8];
#pragma unroll
    for (int r = 0; r < 8; ++r) {
      int qg = q0 + r + hi * 8;
      const unsigned char* mp = mbase + (size_t)qg * S + k0;
      float v0 = mp[n]      ? s0[r] : BIG_NEG;
      float v1 = mp[16 + n] ? s1[r] : BIG_NEG;
      float mx = fmaxf(v0, v1);
      mx = fmaxf(mx, __shfl_xor(mx, 1, 32));
      mx = fmaxf(mx, __shfl_xor(mx, 2, 32));
      mx = fmaxf(mx, __shfl_xor(mx, 4, 32));
      mx = fmaxf(mx, __shfl_xor(mx, 8, 32));
      float mnew = fmaxf(mi[r], mx);
      alpha[r] = __expf(mi[r] - mnew);
      float e0 = __expf(v0 - mnew);
      float e1 = __expf(v1 - mnew);
      float rs = e0 + e1;
      rs += __shfl_xor(rs, 1, 32);
      rs += __shfl_xor(rs, 2, 32);
      rs += __shfl_xor(rs, 4, 32);
      rs += __shfl_xor(rs, 8, 32);
      li[r] = li[r] * alpha[r] + rs;
      mi[r] = mnew;
      p0[r] = e0; p1[r] = e1;
    }

    // ---- stage P (C-layout) into LDS so it can be re-read in A-layout
#pragma unroll
    for (int r = 0; r < 8; ++r) {
      int m = r + hi * 8;
      pl[m * 40 + n]      = (_Float16)p0[r];
      pl[m * 40 + 16 + n] = (_Float16)p1[r];
    }

    // ---- rescale running O by alpha (per output row)
#pragma unroll
    for (int t = 0; t < 8; ++t)
#pragma unroll
      for (int r = 0; r < 8; ++r) o[t][r] *= alpha[r];

    v16h pf = load_a_frag(pl, 40);   // same-wave LDS RAW; DScnt kept in order

    // ---- O += P(16x32) * V(32x128): 8 WMMAs, P-fragment reused
#pragma unroll
    for (int t = 0; t < 8; ++t) {
      const _Float16* vp = vbase + (size_t)(k0 + hi * 8) * D + t * 16 + n;
      v16h vf;
#pragma unroll
      for (int i = 0; i < 8; ++i) { vf[i] = vp[(size_t)i * D]; vf[i + 8] = vp[(size_t)(16 + i) * D]; }
      o[t] = wmma_f16(pf, vf, o[t]);
    }
  }

  // ---- normalize and write [b, q, h*D + d]
  float inv[8];
#pragma unroll
  for (int r = 0; r < 8; ++r) inv[r] = 1.0f / li[r];
#pragma unroll
  for (int t = 0; t < 8; ++t)
#pragma unroll
    for (int r = 0; r < 8; ++r) {
      int qg = q0 + r + hi * 8;
      out[((size_t)b * S + qg) * E + h * D + t * 16 + n] =
          (_Float16)(o[t][r] * inv[r]);
    }
}

// ---------------------------------------------------------------------------
// Host-side launch
// ---------------------------------------------------------------------------
extern "C" void kernel_launch(void* const* d_in, const int* in_sizes, int n_in,
                              void* d_out, int out_size, void* d_ws, size_t ws_size,
                              hipStream_t stream) {
  const float* q      = (const float*)d_in[0];
  const float* kv     = (const float*)d_in[1];
  const float* q_pos  = (const float*)d_in[2];
  const float* kv_pos = (const float*)d_in[3];
  const unsigned char* mask = (const unsigned char*)d_in[4];  // jnp bool = 1 byte
  const float* wq     = (const float*)d_in[5];
  const float* wk     = (const float*)d_in[6];
  const float* wv     = (const float*)d_in[7];
  const float* q_ln   = (const float*)d_in[8];
  const float* k_ln   = (const float*)d_in[9];
  const float* wo     = (const float*)d_in[10];
  float* out          = (float*)d_out;

  const size_t SE = (size_t)Bn * S * E;   // 8,388,608 elements
  const size_t EE = (size_t)E * E;        // 4,194,304 elements

  size_t cur = 0;
  auto alloc = [&](size_t bytes) {
    void* p = (char*)d_ws + cur;
    cur += (bytes + 255) & ~(size_t)255;
    return p;
  };
  _Float16* qi16  = (_Float16*)alloc(SE * 2);
  _Float16* ki16  = (_Float16*)alloc(SE * 2);
  _Float16* vi16  = (_Float16*)alloc(SE * 2);
  _Float16* wq16  = (_Float16*)alloc(EE * 2);
  _Float16* wk16  = (_Float16*)alloc(EE * 2);
  _Float16* wv16  = (_Float16*)alloc(EE * 2);
  _Float16* wo16  = (_Float16*)alloc(EE * 2);
  float*    Qh32  = (float*)alloc(SE * 4);
  float*    Kh32  = (float*)alloc(SE * 4);
  float*    Vh32  = (float*)alloc(SE * 4);
  _Float16* qn16  = (_Float16*)alloc(SE * 2);  // [B,H,S,D]
  _Float16* ktn16 = (_Float16*)alloc(SE * 2);  // [B,H,D,S]
  _Float16* vn16  = (_Float16*)alloc(SE * 2);  // [B,H,S,D]
  _Float16* at16  = (_Float16*)alloc(SE * 2);  // [B,S,H*D]

  // 1. input prep + weight conversion
  prep_inputs<<<(int)(SE / 256), 256, 0, stream>>>(q, kv, q_pos, kv_pos,
                                                   qi16, ki16, vi16);
  f32_to_f16<<<(int)(EE / 256), 256, 0, stream>>>(wq, wq16);
  f32_to_f16<<<(int)(EE / 256), 256, 0, stream>>>(wk, wk16);
  f32_to_f16<<<(int)(EE / 256), 256, 0, stream>>>(wv, wv16);
  f32_to_f16<<<(int)(EE / 256), 256, 0, stream>>>(wo, wo16);

  // 2. projections: [4096 x 2048] x [2048 x 2048], block tile 128x128
  const int M = Bn * S, N = E, K = E;
  dim3 ggrid(M / 128, N / 128);
  gemm_f16f32<<<ggrid, 256, 0, stream>>>(qi16, wq16, Qh32, M, N, K);
  gemm_f16f32<<<ggrid, 256, 0, stream>>>(ki16, wk16, Kh32, M, N, K);
  gemm_f16f32<<<ggrid, 256, 0, stream>>>(vi16, wv16, Vh32, M, N, K);

  // 3. per-head LayerNorm (+ q scaling folded; K written transposed)
  const int ln_blocks = Bn * S * H / 4;
  ln_head<<<ln_blocks, 128, 0, stream>>>(Qh32, q_ln, qn16, INV_SQRT_D, 0);
  ln_head<<<ln_blocks, 128, 0, stream>>>(Kh32, k_ln, ktn16, 1.0f, 1);
  permute_v<<<(int)(SE / 256), 256, 0, stream>>>(Vh32, vn16);

  // 4. flash attention: B*H * (QS/64) blocks, 4 waves each
  flash_attn<<<Bn * H * (S / 64), 128, 0, stream>>>(qn16, ktn16, vn16, mask, at16);

  // 5. output projection -> d_out f32
  gemm_f16f32<<<ggrid, 256, 0, stream>>>(at16, wo16, out, M, N, K);

  (void)in_sizes; (void)n_in; (void)out_size; (void)ws_size;
}
